// MOL_GNN_80161269613396
// MI455X (gfx1250) — matrix operand order repaired and checked by reading, not compiled, not run on hardware
//
#include <hip/hip_runtime.h>
#include <hip/hip_bf16.h>
#include <math.h>

#define Ntot 32768
#define Bsz  1024
#define NPG  32
#define Etot 131072
#define INDIM 74
#define INPAD 96          // INDIM padded to multiple of 32
#define EMB  256
#define Lseq 512
#define H2   512

typedef __attribute__((ext_vector_type(16))) _Float16 v16h;
typedef __attribute__((ext_vector_type(8)))  float    v8f;

#define WMMA16(a,b,c) __builtin_amdgcn_wmma_f32_16x16x32_f16(false,(a),false,(b),(short)0,(c),false,false)

// ---------------------------------------------------------------------------
// A-fragment: convert 32 f32 (two contiguous 8-float runs per lane) -> v16h.
// Layout (16-bit A 16x32, ISA 7.12.2): lane l -> row M=l&15, khalf=l>>4;
// lane base = row*lda + k0 + 8*khalf; halves at +0..7 and +16..23.
// ---------------------------------------------------------------------------
__device__ __forceinline__ v16h cvt_a(const float* __restrict__ p) {
  const float4 c0 = *(const float4*)(p + 0);
  const float4 c1 = *(const float4*)(p + 4);
  const float4 c2 = *(const float4*)(p + 16);
  const float4 c3 = *(const float4*)(p + 20);
  v16h a;
  a[0]  = (_Float16)c0.x; a[1]  = (_Float16)c0.y; a[2]  = (_Float16)c0.z; a[3]  = (_Float16)c0.w;
  a[4]  = (_Float16)c1.x; a[5]  = (_Float16)c1.y; a[6]  = (_Float16)c1.z; a[7]  = (_Float16)c1.w;
  a[8]  = (_Float16)c2.x; a[9]  = (_Float16)c2.y; a[10] = (_Float16)c2.z; a[11] = (_Float16)c2.w;
  a[12] = (_Float16)c3.x; a[13] = (_Float16)c3.y; a[14] = (_Float16)c3.z; a[15] = (_Float16)c3.w;
  return a;
}

__device__ __forceinline__ v16h load_a_frag(const float* __restrict__ A, int lda,
                                            int m0, int k0) {
  const int lane = threadIdx.x & 31;
  return cvt_a(A + (size_t)(m0 + (lane & 15)) * lda + k0 + 8 * (lane >> 4));
}

__device__ __forceinline__ v16h gather_b(const _Float16* __restrict__ p, int ldb) {
  v16h b;
#pragma unroll
  for (int s = 0; s < 8; ++s) {
    b[s]     = p[(size_t)s * ldb];
    b[8 + s] = p[(size_t)(16 + s) * ldb];
  }
  return b;
}

// B layout (32x16): lane l -> column N=l&15, K rows k0+8*(l>>4)+0..7 and +16..23.
__device__ __forceinline__ v16h load_b_frag(const _Float16* __restrict__ W, int ldb,
                                            int k0, int n0) {
  const int lane = threadIdx.x & 31;
  return gather_b(W + (size_t)(k0 + 8 * (lane >> 4)) * ldb + n0 + (lane & 15), ldb);
}

// ---------------------------------------------------------------------------
// Weight conversion: f32 [K,Ncols] -> f16 [Kpad,Ncols], zero rows >= K
// ---------------------------------------------------------------------------
__global__ void w2h_kernel(const float* __restrict__ in, _Float16* __restrict__ out,
                           int K, int Kpad, int Ncols) {
  int idx = blockIdx.x * blockDim.x + threadIdx.x;
  if (idx >= Kpad * Ncols) return;
  int k = idx / Ncols;
  out[idx] = (k < K) ? (_Float16)in[idx] : (_Float16)0.f;
}

// ---------------------------------------------------------------------------
// Generic GEMM: C[M,Ncols] = act(A_f32[M,K] @ W_f16[K,Ncols] + bias)
// 256 threads = 8 waves; wave computes a 16x64 tile, software-pipelined k-loop.
// K % 32 == 0, Ncols % 64 == 0, M % 16 == 0.
// ---------------------------------------------------------------------------
__global__ void wmma_gemm_kernel(const float* __restrict__ A, int lda,
                                 const _Float16* __restrict__ W, int ldb,
                                 const float* __restrict__ bias,
                                 float* __restrict__ C, int ldc,
                                 int M, int K, int Ncols, int act) {
  const int wave = threadIdx.x >> 5;
  const int lane = threadIdx.x & 31;
  const int m0 = (blockIdx.x * 8 + wave) * 16;
  const int n0 = blockIdx.y * 64;
  if (m0 >= M) return;

  const float*    pa = A + (size_t)(m0 + (lane & 15)) * lda + 8 * (lane >> 4);
  const _Float16* pb = W + (size_t)(8 * (lane >> 4)) * ldb + n0 + (lane & 15);
  const size_t bstep = (size_t)32 * ldb;

  v8f acc0 = {}, acc1 = {}, acc2 = {}, acc3 = {};

  // prologue: fragments for k0 = 0
  v16h a  = cvt_a(pa);
  v16h b0 = gather_b(pb, ldb);
  v16h b1 = gather_b(pb + 16, ldb);
  v16h b2 = gather_b(pb + 32, ldb);
  v16h b3 = gather_b(pb + 48, ldb);

  for (int k0 = 32; k0 < K; k0 += 32) {
    pa += 32;
    pb += bstep;
    // issue next-step loads before consuming current fragments
    v16h an = cvt_a(pa);
    v16h c0 = gather_b(pb, ldb);
    v16h c1 = gather_b(pb + 16, ldb);
    v16h c2 = gather_b(pb + 32, ldb);
    v16h c3 = gather_b(pb + 48, ldb);
    acc0 = WMMA16(a, b0, acc0);
    acc1 = WMMA16(a, b1, acc1);
    acc2 = WMMA16(a, b2, acc2);
    acc3 = WMMA16(a, b3, acc3);
    a = an; b0 = c0; b1 = c1; b2 = c2; b3 = c3;
  }
  acc0 = WMMA16(a, b0, acc0);
  acc1 = WMMA16(a, b1, acc1);
  acc2 = WMMA16(a, b2, acc2);
  acc3 = WMMA16(a, b3, acc3);

  const int nn = lane & 15;
  const int mh = lane >> 4;
#pragma unroll
  for (int t = 0; t < 4; ++t) {
    const v8f acc = (t == 0) ? acc0 : (t == 1) ? acc1 : (t == 2) ? acc2 : acc3;
    const int n = n0 + 16 * t + nn;
    const float bv = bias[n];
#pragma unroll
    for (int j = 0; j < 8; ++j) {
      const int m = m0 + 8 * mh + j;
      float v = acc[j] + bv;
      if (act) v = fmaxf(v, 0.f);
      C[(size_t)m * ldc + n] = v;
    }
  }
}

// ---------------------------------------------------------------------------
// Graph plumbing
// ---------------------------------------------------------------------------
__global__ void degree_kernel(const int* __restrict__ dst, float* __restrict__ deg) {
  int e = blockIdx.x * blockDim.x + threadIdx.x;
  if (e < Etot) atomicAdd(&deg[dst[e]], 1.0f);
}

__global__ void scatter_add_kernel(const float* __restrict__ x,
                                   const int* __restrict__ src,
                                   const int* __restrict__ dst,
                                   float* __restrict__ agg, int C) {
  int idx = blockIdx.x * blockDim.x + threadIdx.x;
  if (idx >= Etot * C) return;
  int e = idx / C, c = idx - e * C;
  atomicAdd(&agg[(size_t)dst[e] * C + c], x[(size_t)src[e] * C + c]);
}

// xin[N,Cpad] = (agg + x)/(deg+1), zero-padded columns C..Cpad-1
__global__ void gcn_prep_kernel(const float* __restrict__ x,
                                const float* __restrict__ agg,
                                const float* __restrict__ deg,
                                float* __restrict__ xin, int C, int Cpad) {
  int idx = blockIdx.x * blockDim.x + threadIdx.x;
  if (idx >= Ntot * Cpad) return;
  int n = idx / Cpad, c = idx - n * Cpad;
  float v = 0.f;
  if (c < C) v = (agg[(size_t)n * C + c] + x[(size_t)n * C + c]) / (deg[n] + 1.0f);
  xin[idx] = v;
}

__global__ void segmax_kernel(const float* __restrict__ x3, float* __restrict__ h) {
  int idx = blockIdx.x * blockDim.x + threadIdx.x;  // Bsz*EMB
  int b = idx >> 8, c = idx & 255;
  const float* base = x3 + ((size_t)b * NPG) * EMB + c;
  float m = base[0];
#pragma unroll 4
  for (int n = 1; n < NPG; ++n) m = fmaxf(m, base[(size_t)n * EMB]);
  h[idx] = m;
}

__global__ void xs_sum_kernel(const float* __restrict__ xs, float* __restrict__ out) {
  int b = blockIdx.x, c = threadIdx.x;
  const float* p = xs + (size_t)b * Lseq * EMB + c;
  float s = 0.f;
  for (int l = 0; l < Lseq; ++l) s += p[(size_t)l * EMB];
  out[b * EMB + c] = s;
}

// ---------------------------------------------------------------------------
// Fused protein attention (one block of 8 waves per graph):
//   hs[l] = relu(xs[b,l,:] @ wp + wp_b)     (WMMA -> LDS tile, 16 rows)
//   w[l]  = tanh(dot(hs[l], xgq[b]))
//   xs_att[b] += w[l] * hs[l]
// ---------------------------------------------------------------------------
__global__ void attn_prot_kernel(const float* __restrict__ xs,
                                 const _Float16* __restrict__ wp_w,
                                 const float* __restrict__ wp_b,
                                 const float* __restrict__ xgq,
                                 float* __restrict__ xs_att) {
  __shared__ float hs[16][EMB + 4];
  __shared__ float sh_q[EMB];
  __shared__ float sh_w[16];

  const int b    = blockIdx.x;
  const int tid  = threadIdx.x;
  const int wave = tid >> 5;
  const int lane = tid & 31;
  const int n0   = wave * 32;
  const float* xsb = xs + (size_t)b * Lseq * EMB;

  sh_q[tid] = xgq[b * EMB + tid];
  float acc_att = 0.f;   // channel c == tid
  __syncthreads();

  for (int l0 = 0; l0 < Lseq; l0 += 16) {
    v8f acc0 = {}, acc1 = {};
#pragma unroll
    for (int k0 = 0; k0 < EMB; k0 += 32) {
      v16h a  = load_a_frag(xsb, EMB, l0, k0);
      v16h b0 = load_b_frag(wp_w, EMB, k0, n0);
      v16h b1 = load_b_frag(wp_w, EMB, k0, n0 + 16);
      acc0 = WMMA16(a, b0, acc0);
      acc1 = WMMA16(a, b1, acc1);
    }
    {
      const int n  = n0 + (lane & 15);
      const int mh = lane >> 4;
      const float bv0 = wp_b[n], bv1 = wp_b[n + 16];
#pragma unroll
      for (int j = 0; j < 8; ++j) {
        const int r = 8 * mh + j;
        hs[r][n]      = fmaxf(acc0[j] + bv0, 0.f);
        hs[r][n + 16] = fmaxf(acc1[j] + bv1, 0.f);
      }
    }
    __syncthreads();

    // dot(hs[r], xgq): 16 threads per row, 16 channels each
    {
      const int r = tid >> 4, seg = tid & 15;
      float p = 0.f;
#pragma unroll
      for (int c = 0; c < 16; ++c) p += hs[r][seg * 16 + c] * sh_q[seg * 16 + c];
#pragma unroll
      for (int off = 8; off >= 1; off >>= 1) p += __shfl_xor(p, off, 32);
      if (seg == 0) sh_w[r] = tanhf(p);
    }
    __syncthreads();

#pragma unroll
    for (int r = 0; r < 16; ++r) acc_att += sh_w[r] * hs[r][tid];
    __syncthreads();
  }
  xs_att[b * EMB + tid] = acc_att;
}

// ---------------------------------------------------------------------------
// Compound-side attention
// ---------------------------------------------------------------------------
__global__ void comp_attn_kernel(const float* __restrict__ xgr,
                                 const float* __restrict__ xs_sum,
                                 float* __restrict__ xg_attn) {
  __shared__ float sh_s[EMB];
  __shared__ float sh_w[NPG];
  const int b = blockIdx.x, tid = threadIdx.x;
  sh_s[tid] = xs_sum[b * EMB + tid];
  __syncthreads();

  {
    const int n = tid >> 3, seg = tid & 7;   // 32 nodes x 8 threads
    const float* xr = xgr + ((size_t)b * NPG + n) * EMB;
    float p = 0.f;
#pragma unroll
    for (int c = 0; c < 32; ++c) p += xr[seg * 32 + c] * sh_s[seg * 32 + c];
#pragma unroll
    for (int off = 4; off >= 1; off >>= 1) p += __shfl_xor(p, off, 32);
    if (seg == 0) sh_w[n] = tanhf(p);
  }
  __syncthreads();

  const float* base = xgr + (size_t)b * NPG * EMB;
  float m = -INFINITY;
#pragma unroll 4
  for (int n = 0; n < NPG; ++n) m = fmaxf(m, base[(size_t)n * EMB + tid] * sh_w[n]);
  xg_attn[b * EMB + tid] = m;
}

// ---------------------------------------------------------------------------
// BatchNorm statistics + output head
// ---------------------------------------------------------------------------
__global__ void bn_stats_kernel(const float* __restrict__ v,
                                float* __restrict__ mean, float* __restrict__ rstd) {
  const int c = blockIdx.x, tid = threadIdx.x;
  float s = 0.f, s2 = 0.f;
  for (int b = tid; b < Bsz; b += 256) {
    float x = v[(size_t)b * EMB + c];
    s += x; s2 += x * x;
  }
  __shared__ float sh1[256], sh2[256];
  sh1[tid] = s; sh2[tid] = s2; __syncthreads();
  for (int off = 128; off >= 1; off >>= 1) {
    if (tid < off) { sh1[tid] += sh1[tid + off]; sh2[tid] += sh2[tid + off]; }
    __syncthreads();
  }
  if (tid == 0) {
    float m = sh1[0] / (float)Bsz;
    float var = sh2[0] / (float)Bsz - m * m;
    mean[c] = m; rstd[c] = rsqrtf(var + 1e-5f);
  }
}

__global__ void final_kernel(const float* __restrict__ comp, const float* __restrict__ prot,
                             const float* __restrict__ mc, const float* __restrict__ rc,
                             const float* __restrict__ mp, const float* __restrict__ rp,
                             const float* __restrict__ gc, const float* __restrict__ bc,
                             const float* __restrict__ gp, const float* __restrict__ bp,
                             const float* __restrict__ wout, const float* __restrict__ bout,
                             float* __restrict__ out) {
  int b = blockIdx.x * blockDim.x + threadIdx.x;
  if (b >= Bsz) return;
  float a0 = bout[0], a1 = bout[1];
  for (int c = 0; c < EMB; ++c) {
    float nv = (comp[(size_t)b * EMB + c] - mc[c]) * rc[c] * gc[c] + bc[c];
    a0 += nv * wout[c * 2 + 0];
    a1 += nv * wout[c * 2 + 1];
    float pv = (prot[(size_t)b * EMB + c] - mp[c]) * rp[c] * gp[c] + bp[c];
    a0 += pv * wout[(EMB + c) * 2 + 0];
    a1 += pv * wout[(EMB + c) * 2 + 1];
  }
  out[b * 2 + 0] = a0;
  out[b * 2 + 1] = a1;
}

// ---------------------------------------------------------------------------
// Host side
// ---------------------------------------------------------------------------
static void gemm(hipStream_t s, const float* A, int lda, const _Float16* W, int ldb,
                 const float* bias, float* C, int ldc, int M, int K, int Nc, int act) {
  dim3 grid((M + 127) / 128, Nc / 64);
  wmma_gemm_kernel<<<grid, 256, 0, s>>>(A, lda, W, ldb, bias, C, ldc, M, K, Nc, act);
}

static void w2h(hipStream_t s, const float* in, _Float16* out, int K, int Kpad, int Nc) {
  w2h_kernel<<<(Kpad * Nc + 255) / 256, 256, 0, s>>>(in, out, K, Kpad, Nc);
}

extern "C" void kernel_launch(void* const* d_in, const int* in_sizes, int n_in,
                              void* d_out, int out_size, void* d_ws, size_t ws_size,
                              hipStream_t stream) {
  const float* f     = (const float*)d_in[0];
  const float* xs    = (const float*)d_in[1];
  const int*   src   = (const int*)  d_in[2];
  const int*   dst   = (const int*)  d_in[3];
  const float* gw0   = (const float*)d_in[4];  const float* gb0 = (const float*)d_in[5];
  const float* gw1   = (const float*)d_in[6];  const float* gb1 = (const float*)d_in[7];
  const float* gw2   = (const float*)d_in[8];  const float* gb2 = (const float*)d_in[9];
  const float* mp0w  = (const float*)d_in[10]; const float* mp0b = (const float*)d_in[11];
  const float* mp1w  = (const float*)d_in[12]; const float* mp1b = (const float*)d_in[13];
  const float* mg0w  = (const float*)d_in[14]; const float* mg0b = (const float*)d_in[15];
  const float* mg1w  = (const float*)d_in[16]; const float* mg1b = (const float*)d_in[17];
  const float* fpw   = (const float*)d_in[18]; const float* fpb  = (const float*)d_in[19];
  const float* fgw   = (const float*)d_in[20]; const float* fgb  = (const float*)d_in[21];
  const float* wpw   = (const float*)d_in[22]; const float* wpb  = (const float*)d_in[23];
  const float* wcw   = (const float*)d_in[24]; const float* wcb  = (const float*)d_in[25];
  const float* wcgw  = (const float*)d_in[26]; const float* wcgb = (const float*)d_in[27];
  const float* woutw = (const float*)d_in[28]; const float* woutb= (const float*)d_in[29];
  const float* bncg  = (const float*)d_in[30]; const float* bncb = (const float*)d_in[31];
  const float* bnpg  = (const float*)d_in[32]; const float* bnpb = (const float*)d_in[33];
  float* out = (float*)d_out;

  // ---- workspace carve-up: f32 region then f16 weight pool (32B-aligned) ----
  float* ws = (float*)d_ws;
  size_t o = 0;
  float* deg = ws + o; o += Ntot;
  float* P   = ws + o; o += (size_t)Ntot * 128;   // gcn prep input (padded)
  float* G   = ws + o; o += (size_t)Ntot * 128;   // aggregation
  float* Abuf= ws + o; o += (size_t)Ntot * H2;    // ping
  float* Bbuf= ws + o; o += (size_t)Ntot * H2;    // pong
  float* Cbuf= ws + o; o += (size_t)Ntot * EMB;   // x3
  float* H0  = ws + o; o += (size_t)Bsz * EMB;    // pooled
  float* H1  = ws + o; o += (size_t)Bsz * H2;
  float* H2b = ws + o; o += (size_t)Bsz * H2;
  float* H3  = ws + o; o += (size_t)Bsz * EMB;    // xg
  float* H4  = ws + o; o += (size_t)Bsz * EMB;    // xg_q
  float* S   = ws + o; o += (size_t)Bsz * EMB;    // xs_sum
  float* T   = ws + o; o += (size_t)Bsz * EMB;    // xs_att
  float* U   = ws + o; o += (size_t)Bsz * EMB;    // xg_attn
  float* MC  = ws + o; o += EMB; float* RC = ws + o; o += EMB;
  float* MP  = ws + o; o += EMB; float* RP = ws + o; o += EMB;

  _Float16* wh = (_Float16*)(ws + o);
  size_t q = 0;
  _Float16* gw0h = wh + q; q += (size_t)INPAD * 128;
  _Float16* gw1h = wh + q; q += (size_t)128 * 128;
  _Float16* gw2h = wh + q; q += (size_t)128 * EMB;
  _Float16* mp0h = wh + q; q += (size_t)EMB * H2;
  _Float16* mp1h = wh + q; q += (size_t)H2 * H2;
  _Float16* mg0h = wh + q; q += (size_t)EMB * H2;
  _Float16* mg1h = wh + q; q += (size_t)H2 * H2;
  _Float16* fph  = wh + q; q += (size_t)H2 * EMB;
  _Float16* fgh  = wh + q; q += (size_t)H2 * EMB;
  _Float16* wph  = wh + q; q += (size_t)EMB * EMB;
  _Float16* wch  = wh + q; q += (size_t)EMB * EMB;
  _Float16* wcgh = wh + q; q += (size_t)EMB * EMB;

  // ---- convert weights to f16 (gcn_w0 zero-padded to 96 rows) ----
  w2h(stream, gw0, gw0h, INDIM, INPAD, 128);
  w2h(stream, gw1, gw1h, 128, 128, 128);
  w2h(stream, gw2, gw2h, 128, 128, EMB);
  w2h(stream, mp0w, mp0h, EMB, EMB, H2);
  w2h(stream, mp1w, mp1h, H2, H2, H2);
  w2h(stream, mg0w, mg0h, EMB, EMB, H2);
  w2h(stream, mg1w, mg1h, H2, H2, H2);
  w2h(stream, fpw,  fph,  H2, H2, EMB);
  w2h(stream, fgw,  fgh,  H2, H2, EMB);
  w2h(stream, wpw,  wph,  EMB, EMB, EMB);
  w2h(stream, wcw,  wch,  EMB, EMB, EMB);
  w2h(stream, wcgw, wcgh, EMB, EMB, EMB);

  // ---- degree ----
  hipMemsetAsync(deg, 0, Ntot * sizeof(float), stream);
  degree_kernel<<<(Etot + 255) / 256, 256, 0, stream>>>(dst, deg);

  // --- GCN layer 0: f[N,74] -> x1 = Abuf[N,128]  (K padded to 96)
  hipMemsetAsync(G, 0, (size_t)Ntot * INDIM * sizeof(float), stream);
  scatter_add_kernel<<<(Etot * INDIM + 255) / 256, 256, 0, stream>>>(f, src, dst, G, INDIM);
  gcn_prep_kernel<<<(Ntot * INPAD + 255) / 256, 256, 0, stream>>>(f, G, deg, P, INDIM, INPAD);
  gemm(stream, P, INPAD, gw0h, 128, gb0, Abuf, 128, Ntot, INPAD, 128, 1);

  // --- GCN layer 1: x1 -> x2 = Bbuf[N,128]
  hipMemsetAsync(G, 0, (size_t)Ntot * 128 * sizeof(float), stream);
  scatter_add_kernel<<<(Etot * 128 + 255) / 256, 256, 0, stream>>>(Abuf, src, dst, G, 128);
  gcn_prep_kernel<<<(Ntot * 128 + 255) / 256, 256, 0, stream>>>(Abuf, G, deg, P, 128, 128);
  gemm(stream, P, 128, gw1h, 128, gb1, Bbuf, 128, Ntot, 128, 128, 1);

  // --- GCN layer 2: x2 -> x3 = Cbuf[N,256]
  hipMemsetAsync(G, 0, (size_t)Ntot * 128 * sizeof(float), stream);
  scatter_add_kernel<<<(Etot * 128 + 255) / 256, 256, 0, stream>>>(Bbuf, src, dst, G, 128);
  gcn_prep_kernel<<<(Ntot * 128 + 255) / 256, 256, 0, stream>>>(Bbuf, G, deg, P, 128, 128);
  gemm(stream, P, 128, gw2h, EMB, gb2, Cbuf, EMB, Ntot, 128, EMB, 1);

  // --- pooled graph features
  segmax_kernel<<<(Bsz * EMB) / 256, 256, 0, stream>>>(Cbuf, H0);

  // --- fn path (per-node): x3 -> 512 -> 512 -> 256 -> xgr(256)
  gemm(stream, Cbuf, EMB, mg0h, H2, mg0b, Abuf, H2, Ntot, EMB, H2, 0);   // fnA
  gemm(stream, Abuf, H2, mg1h, H2, mg1b, Bbuf, H2, Ntot, H2, H2, 0);     // fnB
  gemm(stream, Bbuf, H2, fgh, EMB, fgb, Abuf, EMB, Ntot, H2, EMB, 1);    // fn (reuse A)
  gemm(stream, Abuf, EMB, wcgh, EMB, wcgb, Bbuf, EMB, Ntot, EMB, EMB, 1);// xgr (reuse B)

  // --- h path (per-graph)
  gemm(stream, H0, EMB, mp0h, H2, mp0b, H1, H2, Bsz, EMB, H2, 0);
  gemm(stream, H1, H2, mp1h, H2, mp1b, H2b, H2, Bsz, H2, H2, 0);
  gemm(stream, H2b, H2, fph, EMB, fpb, H3, EMB, Bsz, H2, EMB, 1);        // xg
  gemm(stream, H3, EMB, wch, EMB, wcb, H4, EMB, Bsz, EMB, EMB, 1);       // xg_q

  // --- protein side
  xs_sum_kernel<<<Bsz, EMB, 0, stream>>>(xs, S);
  attn_prot_kernel<<<Bsz, 256, 0, stream>>>(xs, wph, wpb, H4, T);

  // --- compound side
  comp_attn_kernel<<<Bsz, 256, 0, stream>>>(Bbuf, S, U);

  // --- BN + output head
  bn_stats_kernel<<<EMB, 256, 0, stream>>>(U, MC, RC);
  bn_stats_kernel<<<EMB, 256, 0, stream>>>(T, MP, RP);
  final_kernel<<<(Bsz + 255) / 256, 256, 0, stream>>>(U, T, MC, RC, MP, RP,
                                                      bncg, bncb, bnpg, bnpb,
                                                      woutw, woutb, out);
  (void)in_sizes; (void)n_in; (void)out_size; (void)ws_size;
}
